// PointNetSetAbstraction_687194767483
// MI455X (gfx1250) — matrix operand order repaired and checked by reading, not compiled, not run on hardware
//
#include <hip/hip_runtime.h>
#include <math.h>

typedef __attribute__((ext_vector_type(16))) __bf16 v16bf;
typedef __attribute__((ext_vector_type(8)))  __bf16 v8bf;
typedef __attribute__((ext_vector_type(8)))  float  v8f;

#define NPOINT   1024
#define NIN      4096
#define BATCH    16
#define NSAMPLE  32
#define RADIUS2  0.04f
#define BN_EPS   1e-5f
#define MTOT     (BATCH * NPOINT * NSAMPLE)   // 524288

static __device__ __forceinline__ __bf16 f2bf(float f) { return (__bf16)f; }

// ---------------------------------------------------------------------------
// Kernel 1: farthest point sampling, one block per batch; writes new_xyz into
// d_out[0 .. 16*1024*3). xyz + running distances live in LDS.
// ---------------------------------------------------------------------------
__global__ __launch_bounds__(256) void fps_kernel(const float* __restrict__ xyz,
                                                  float* __restrict__ out_newxyz) {
  const int b = blockIdx.x;
  const int tid = threadIdx.x;
  __shared__ float sx[NIN], sy[NIN], sz[NIN], sd[NIN];
  __shared__ float swv[8];
  __shared__ int   swi[8];
  __shared__ int   sfar;

  const float* base = xyz + (size_t)b * NIN * 3;
  for (int i = tid; i < NIN; i += 256) {
    sx[i] = base[i * 3 + 0];
    sy[i] = base[i * 3 + 1];
    sz[i] = base[i * 3 + 2];
    sd[i] = 1e10f;
  }
  __syncthreads();

  int far = 0;
  for (int it = 0; it < NPOINT; ++it) {
    const float px = sx[far], py = sy[far], pz = sz[far];
    if (tid == 0) {
      size_t o = ((size_t)b * NPOINT + it) * 3;
      out_newxyz[o + 0] = px;
      out_newxyz[o + 1] = py;
      out_newxyz[o + 2] = pz;
    }
    float bestv = -1.0f;
    int   besti = 0;
    for (int i = tid; i < NIN; i += 256) {
      float dx = sx[i] - px, dy = sy[i] - py, dz = sz[i] - pz;
      float d = dx * dx + dy * dy + dz * dz;
      float nd = fminf(sd[i], d);
      sd[i] = nd;
      if (nd > bestv) { bestv = nd; besti = i; }
    }
    // wave32 reduction, tie-break on lowest index (matches jnp.argmax)
    for (int off = 16; off > 0; off >>= 1) {
      float ov = __shfl_xor(bestv, off, 32);
      int   oi = __shfl_xor(besti, off, 32);
      if (ov > bestv || (ov == bestv && oi < besti)) { bestv = ov; besti = oi; }
    }
    if ((tid & 31) == 0) { swv[tid >> 5] = bestv; swi[tid >> 5] = besti; }
    __syncthreads();
    if (tid == 0) {
      float bv = swv[0]; int bi = swi[0];
      for (int j = 1; j < 8; ++j)
        if (swv[j] > bv || (swv[j] == bv && swi[j] < bi)) { bv = swv[j]; bi = swi[j]; }
      sfar = bi;
    }
    __syncthreads();
    far = sfar;
  }
}

// ---------------------------------------------------------------------------
// Kernel 2: ball query (first 32 in-radius indices, ascending) + gather +
// concat -> x0 (M x 96 f32, channels: 3 xyz_norm, 64 points, 29 zero pad).
// One wave per query point.
// ---------------------------------------------------------------------------
__global__ __launch_bounds__(256) void ballquery_build_kernel(
    const float* __restrict__ xyz, const float* __restrict__ points,
    const float* __restrict__ newxyz, float* __restrict__ x0) {
  const int gwave = (int)((blockIdx.x * blockDim.x + threadIdx.x) >> 5);
  const int lane  = threadIdx.x & 31;
  const int wl    = threadIdx.x >> 5;
  const int b = gwave >> 10;           // / NPOINT
  __shared__ int slots[8][NSAMPLE];

  const float* q = newxyz + (size_t)gwave * 3;
  const float qx = q[0], qy = q[1], qz = q[2];
  const float* xb = xyz + (size_t)b * NIN * 3;

  int cnt = 0;
  for (int basei = 0; basei < NIN && cnt < NSAMPLE; basei += 32) {
    const int i = basei + lane;
    float dx = xb[i * 3 + 0] - qx;
    float dy = xb[i * 3 + 1] - qy;
    float dz = xb[i * 3 + 2] - qz;
    float d = dx * dx + dy * dy + dz * dz;
    bool inr = (d <= RADIUS2);
    unsigned mask = (unsigned)__ballot(inr);
    int pos = cnt + __popc(mask & ((1u << lane) - 1u));
    if (inr && pos < NSAMPLE) slots[wl][pos] = i;
    cnt = min(NSAMPLE, cnt + __popc(mask));
  }
  __syncthreads();

  const int sel = slots[wl][lane < cnt ? lane : 0];  // cnt >= 1 (query is a data point)

  const size_t m = (size_t)gwave * NSAMPLE + lane;
  float* dst = x0 + m * 96;
  dst[0] = xb[sel * 3 + 0] - qx;
  dst[1] = xb[sel * 3 + 1] - qy;
  dst[2] = xb[sel * 3 + 2] - qz;
  const float* pp = points + ((size_t)b * NIN + sel) * 64;
  #pragma unroll 8
  for (int c = 0; c < 64; ++c) dst[3 + c] = pp[c];
  #pragma unroll
  for (int c = 67; c < 96; ++c) dst[c] = 0.0f;
}

// ---------------------------------------------------------------------------
// GEMM + fused BN statistics.  Y[m,o] = sum_c a(m,c) * w(o,c)
// a = A (layer0) or relu(A*scale+shift) (layers 1,2).
// Block: 256 threads = 8 waves, 128 rows per block, each wave does a 16-row
// strip across all N columns with v_wmma_f32_16x16x32_bf16.
// ---------------------------------------------------------------------------
template <int NT, int KP, bool AFFINE>
__global__ __launch_bounds__(256) void gemm_bn_kernel(
    const float* __restrict__ A, const float* __restrict__ scale,
    const float* __restrict__ shift, const float* __restrict__ W, int Cin,
    float* __restrict__ Y, float* __restrict__ gsum, float* __restrict__ gss) {
  constexpr int N = NT * 16;
  __shared__ __attribute__((aligned(16))) __bf16 sW[N * KP];  // w(o,c) bf16, zero padded
  __shared__ float sScale[KP], sShift[KP];
  __shared__ float lsum[N], lss[N];

  const int tid = threadIdx.x;
  for (int idx = tid; idx < N * KP; idx += 256) {
    int o = idx / KP, c = idx % KP;
    sW[idx] = f2bf(c < Cin ? W[o * Cin + c] : 0.0f);
  }
  if (AFFINE) {
    for (int c = tid; c < KP; c += 256) { sScale[c] = scale[c]; sShift[c] = shift[c]; }
  }
  for (int n = tid; n < N; n += 256) { lsum[n] = 0.0f; lss[n] = 0.0f; }
  __syncthreads();

  const int wave = tid >> 5, lane = tid & 31;
  const int hi16 = lane >> 4, l16 = lane & 15;
  const size_t rowA = (size_t)blockIdx.x * 128 + wave * 16 + l16;
  const float* aRow = A + rowA * KP;

  v8f acc[NT] = {};

  #pragma unroll
  for (int kk = 0; kk < KP; kk += 32) {
    const int k1 = kk + hi16 * 8;   // first 8 K values this lane holds
    const int k2 = k1 + 16;         // second 8
    v8bf g1 = {}, g2 = {};
    if (AFFINE) {
      #pragma unroll
      for (int j = 0; j < 8; ++j) {
        float v1 = fmaxf(fmaf(aRow[k1 + j], sScale[k1 + j], sShift[k1 + j]), 0.0f);
        float v2 = fmaxf(fmaf(aRow[k2 + j], sScale[k2 + j], sShift[k2 + j]), 0.0f);
        g1[j] = f2bf(v1);
        g2[j] = f2bf(v2);
      }
    } else {
      #pragma unroll
      for (int j = 0; j < 8; ++j) {
        g1[j] = f2bf(aRow[k1 + j]);
        g2[j] = f2bf(aRow[k2 + j]);
      }
    }
    v16bf afrag = __builtin_shufflevector(g1, g2, 0, 1, 2, 3, 4, 5, 6, 7,
                                          8, 9, 10, 11, 12, 13, 14, 15);
    const int kb = kk + hi16 * 8;   // B fragment: same K order as A (8 at kb, 8 at kb+16)
    #pragma unroll
    for (int t = 0; t < NT; ++t) {
      const __bf16* wp = &sW[(t * 16 + l16) * KP + kb];
      v8bf b1 = *(const v8bf*)wp;
      v8bf b2 = *(const v8bf*)(wp + 16);
      v16bf bfrag = __builtin_shufflevector(b1, b2, 0, 1, 2, 3, 4, 5, 6, 7,
                                            8, 9, 10, 11, 12, 13, 14, 15);
      acc[t] = __builtin_amdgcn_wmma_f32_16x16x32_bf16(
          false, afrag, false, bfrag, (short)0, acc[t], false, false);
    }
  }

  // epilogue: store f32 + per-channel partial stats
  const size_t rowBase = (size_t)blockIdx.x * 128 + wave * 16 + hi16 * 8;
  #pragma unroll
  for (int t = 0; t < NT; ++t) {
    float s = 0.0f, ss = 0.0f;
    #pragma unroll
    for (int r = 0; r < 8; ++r) {
      float v = acc[t][r];
      Y[(rowBase + r) * N + t * 16 + l16] = v;
      s += v;
      ss += v * v;
    }
    atomicAdd(&lsum[t * 16 + l16], s);   // ds_add_f32
    atomicAdd(&lss[t * 16 + l16], ss);
  }
  __syncthreads();
  for (int n = tid; n < N; n += 256) {
    atomicAdd(&gsum[n], lsum[n]);
    atomicAdd(&gss[n], lss[n]);
  }
}

// ---------------------------------------------------------------------------
// Stats finalize: scale = g * rsqrt(var+eps), shift = bt - mean*scale.
// (bias b cancels inside batch norm, so it is never applied.)
// ---------------------------------------------------------------------------
__global__ void finalize_stats_kernel(const float* __restrict__ gsum,
                                      const float* __restrict__ gss,
                                      const float* __restrict__ g,
                                      const float* __restrict__ bt,
                                      float* __restrict__ scale,
                                      float* __restrict__ shift, int N,
                                      float invM) {
  int n = blockIdx.x * blockDim.x + threadIdx.x;
  if (n < N) {
    float mean = gsum[n] * invM;
    float var = gss[n] * invM - mean * mean;
    float s = g[n] * rsqrtf(var + BN_EPS);
    scale[n] = s;
    shift[n] = bt[n] - mean * s;
  }
}

__global__ void zero_kernel(float* __restrict__ p, int n) {
  int i = blockIdx.x * blockDim.x + threadIdx.x;
  if (i < n) p[i] = 0.0f;
}

// ---------------------------------------------------------------------------
// Final: BN + relu + max over the 32 samples -> new_points (B,1024,128)
// ---------------------------------------------------------------------------
__global__ __launch_bounds__(128) void bn_relu_maxpool_kernel(
    const float* __restrict__ Y2, const float* __restrict__ scale,
    const float* __restrict__ shift, float* __restrict__ out) {
  const int bs = blockIdx.x;      // 0..16383
  const int n = threadIdx.x;      // 0..127
  const float sc = scale[n], sh = shift[n];
  const float* p = Y2 + (size_t)bs * NSAMPLE * 128 + n;
  float m = 0.0f;                 // relu output is >= 0
  #pragma unroll 8
  for (int k = 0; k < NSAMPLE; ++k)
    m = fmaxf(m, fmaxf(fmaf(p[k * 128], sc, sh), 0.0f));
  out[(size_t)BATCH * NPOINT * 3 + (size_t)bs * 128 + n] = m;
}

// ---------------------------------------------------------------------------
extern "C" void kernel_launch(void* const* d_in, const int* in_sizes, int n_in,
                              void* d_out, int out_size, void* d_ws, size_t ws_size,
                              hipStream_t stream) {
  const float* xyz    = (const float*)d_in[0];
  const float* points = (const float*)d_in[1];
  const float* w0 = (const float*)d_in[2];
  const float* g0 = (const float*)d_in[4];
  const float* bt0 = (const float*)d_in[5];
  const float* w1 = (const float*)d_in[6];
  const float* g1 = (const float*)d_in[8];
  const float* bt1 = (const float*)d_in[9];
  const float* w2 = (const float*)d_in[10];
  const float* g2 = (const float*)d_in[12];
  const float* bt2 = (const float*)d_in[13];
  float* out = (float*)d_out;

  char* ws = (char*)d_ws;
  const size_t M = MTOT;
  float* x0 = (float*)ws;                          // M x 96
  float* y0 = (float*)(ws + M * 96 * 4);           // M x 64
  float* y1 = (float*)((char*)y0 + M * 64 * 4);    // M x 64
  float* y2 = (float*)((char*)y1 + M * 64 * 4);    // M x 128
  float* stats = (float*)((char*)y2 + M * 128 * 4);
  float* gsum0 = stats + 0,   *gss0 = stats + 64;
  float* gsum1 = stats + 128, *gss1 = stats + 192;
  float* gsum2 = stats + 256, *gss2 = stats + 384;
  float* scale0 = stats + 512, *shift0 = stats + 576;
  float* scale1 = stats + 640, *shift1 = stats + 704;
  float* scale2 = stats + 768, *shift2 = stats + 896;

  const float invM = 1.0f / (float)M;

  zero_kernel<<<1, 512, 0, stream>>>(stats, 512);
  fps_kernel<<<BATCH, 256, 0, stream>>>(xyz, out);
  ballquery_build_kernel<<<(BATCH * NPOINT) / 8, 256, 0, stream>>>(xyz, points, out, x0);

  gemm_bn_kernel<4, 96, false><<<M / 128, 256, 0, stream>>>(
      x0, nullptr, nullptr, w0, 67, y0, gsum0, gss0);
  finalize_stats_kernel<<<1, 64, 0, stream>>>(gsum0, gss0, g0, bt0, scale0, shift0, 64, invM);

  gemm_bn_kernel<4, 64, true><<<M / 128, 256, 0, stream>>>(
      y0, scale0, shift0, w1, 64, y1, gsum1, gss1);
  finalize_stats_kernel<<<1, 64, 0, stream>>>(gsum1, gss1, g1, bt1, scale1, shift1, 64, invM);

  gemm_bn_kernel<8, 64, true><<<M / 128, 256, 0, stream>>>(
      y1, scale1, shift1, w2, 64, y2, gsum2, gss2);
  finalize_stats_kernel<<<1, 128, 0, stream>>>(gsum2, gss2, g2, bt2, scale2, shift2, 128, invM);

  bn_relu_maxpool_kernel<<<BATCH * NPOINT, 128, 0, stream>>>(y2, scale2, shift2, out);
}
